// InfiniteAttention_75866302316755
// MI455X (gfx1250) — matrix operand order repaired and checked by reading, not compiled, not run on hardware
//
#include <hip/hip_runtime.h>

// CDNA5 / gfx1250 implementation of Infini-attention reference.
// Pipeline:
//   0) conv_f16   : x, wq, wk, wv, wo -> f16 (one cheap bandwidth pass)
//   1) gemm_qkv   : Q/K/V = x @ W^T  (f16 WMMA, async-LDS tile staging)
//   2) seg_delta  : per-(b,h,seg) delta_mem = (elu(k)+1)^T @ v, delta_z  (WMMA)
//   3) prefix_mem : exclusive prefix over segments -> old-state mem/z
//   4) attn_seg   : segment softmax attention + linear-memory read (WMMA)
//   5) gemm_out   : out = att @ wo^T + b  (f16 WMMA, async-LDS tile staging)

typedef __attribute__((ext_vector_type(16))) _Float16 v16h;
typedef __attribute__((ext_vector_type(8)))  _Float16 v8h;
typedef __attribute__((ext_vector_type(4)))  _Float16 v4h;
typedef __attribute__((ext_vector_type(8)))  float    v8f;
typedef __attribute__((ext_vector_type(4)))  float    v4f;
typedef __attribute__((ext_vector_type(4)))  int      v4i;

#define B_   4
#define L_   8192
#define D_   1024
#define H_   16
#define DK_  64
#define SEG  512
#define NSEG 16

#ifndef __has_builtin
#define __has_builtin(x) 0
#endif

// ---- CDNA5 async global->LDS copy (ASYNCcnt path), 16 bytes per lane ----
// Probe-revealed builtin signature: (int4 addrspace(1)*, int4 addrspace(3)*,
// imm int, imm int). Cast through integers to synthesize the AS pointers.
__device__ __forceinline__ void async_g2l_16B(const void* g, const void* l) {
#if __has_builtin(__builtin_amdgcn_global_load_async_to_lds_b128)
  typedef __attribute__((address_space(1))) v4i* gp1_t;
  typedef __attribute__((address_space(3))) v4i* lp3_t;
  __builtin_amdgcn_global_load_async_to_lds_b128(
      (gp1_t)(unsigned long long)(size_t)g, (lp3_t)(unsigned)(size_t)l, 0, 0);
#else
  unsigned long long ga = (unsigned long long)(size_t)g;
  unsigned la = (unsigned)(size_t)l;  // flat LDS addr[31:0] == wave LDS offset
  asm volatile("global_load_async_to_lds_b128 %0, %1, off"
               :: "v"(la), "v"(ga) : "memory");
#endif
}

__device__ __forceinline__ void wait_async0() {
#if __has_builtin(__builtin_amdgcn_s_wait_asynccnt)
  __builtin_amdgcn_s_wait_asynccnt(0);
#else
  asm volatile("s_wait_asynccnt 0x0" ::: "memory");
#endif
}

__device__ __forceinline__ v8f wmma_f16f32(v16h a, v16h b, v8f c) {
  // D = A(16x32 f16) * B(32x16 f16) + C(16x16 f32)
  return __builtin_amdgcn_wmma_f32_16x16x32_f16(false, a, false, b, (short)0, c,
                                                false, false);
}

// A-matrix 16x32 f16 fragment from LDS tile origin (row stride ld halves,
// ld must be a multiple of 8). Lane L: row = L%16; halves K = g*8..g*8+7 and
// K = g*8+16..g*8+23 where g = L/16  (ISA 7.12.2 16-bit A layout).
__device__ __forceinline__ v16h ld_a_frag(const _Float16* tile, int ld, int lane) {
  const _Float16* p = tile + (size_t)(lane & 15) * ld + ((lane >> 4) << 3);
  v16h r;
  *(v8h*)&r       = *(const v8h*)p;
  *((v8h*)&r + 1) = *(const v8h*)(p + 16);
  return r;
}

// B-matrix 32x16 f16 fragment; LDS holds B as [col][k] (column-major of B).
// Lane L: col = L%16; K = g*8..g*8+7 and K = g*8+16..g*8+23 where g = L/16.
__device__ __forceinline__ v16h ld_b_frag(const _Float16* tile, int ld, int lane) {
  const _Float16* p = tile + (size_t)(lane & 15) * ld + ((lane >> 4) << 3);
  v16h r;
  *(v8h*)&r       = *(const v8h*)p;
  *((v8h*)&r + 1) = *(const v8h*)(p + 16);
  return r;
}

// ---------------------------------------------------------------- kernel 0
__global__ __launch_bounds__(256) void conv_f16_kernel(
    const float* __restrict__ src, _Float16* __restrict__ dst, int n4)
{
  int i = blockIdx.x * blockDim.x + threadIdx.x;
  int stride = gridDim.x * blockDim.x;
  for (; i < n4; i += stride) {
    v4f v = *(const v4f*)(src + (size_t)i * 4);
    v4h o;
#pragma unroll
    for (int j = 0; j < 4; j++) o[j] = (_Float16)v[j];
    *(v4h*)(dst + (size_t)i * 4) = o;
  }
}

// ---------------------------------------------------------------- GEMM tiles
#define GTM 128
#define GTN 128
#define GTK 32
#define GLDK 40   // GTK + 8 halves pad

// ---------------------------------------------------------------- kernel 1
__global__ __launch_bounds__(256) void gemm_qkv_kernel(
    const _Float16* __restrict__ Xh,
    const _Float16* __restrict__ Wq, const _Float16* __restrict__ Wk,
    const _Float16* __restrict__ Wv,
    _Float16* __restrict__ Qh, _Float16* __restrict__ Kh,
    _Float16* __restrict__ Vh)
{
  __shared__ _Float16 As[GTM][GLDK];
  __shared__ _Float16 Bs[GTN][GLDK];
  const _Float16* W = (blockIdx.z == 0) ? Wq : (blockIdx.z == 1) ? Wk : Wv;
  _Float16*      P = (blockIdx.z == 0) ? Qh : (blockIdx.z == 1) ? Kh : Vh;
  const int N = D_, K = D_;
  const int m0 = blockIdx.y * GTM, n0 = blockIdx.x * GTN;
  const int tid = threadIdx.x, lane = tid & 31, wid = tid >> 5;
  const int wm = (wid & 3) * 32, wn = (wid >> 2) * 64;

  const v8f vz = {0.f, 0.f, 0.f, 0.f, 0.f, 0.f, 0.f, 0.f};
  v8f acc[2][4];
  for (int i = 0; i < 2; i++)
    for (int j = 0; j < 4; j++) acc[i][j] = vz;

  for (int k0 = 0; k0 < K; k0 += GTK) {
#pragma unroll
    for (int i = 0; i < 2; i++) {   // 128x32 f16 tiles: 512 x 16B each
      int idx = tid + (i << 8);
      int r = idx >> 2, c = (idx & 3) << 3;
      async_g2l_16B(Xh + (size_t)(m0 + r) * K + k0 + c, &As[r][c]);
      async_g2l_16B(W  + (size_t)(n0 + r) * K + k0 + c, &Bs[r][c]);
    }
    if (k0 + GTK < K)   // hint next A tile into cache (global_prefetch_b8)
      __builtin_prefetch(Xh + (size_t)(m0 + (tid >> 1)) * K + k0 + GTK, 0, 1);
    wait_async0();
    __syncthreads();

    v16h af[2], bf[4];
#pragma unroll
    for (int mi = 0; mi < 2; mi++)
      af[mi] = ld_a_frag(&As[wm + mi * 16][0], GLDK, lane);
#pragma unroll
    for (int ni = 0; ni < 4; ni++)
      bf[ni] = ld_b_frag(&Bs[wn + ni * 16][0], GLDK, lane);
#pragma unroll
    for (int mi = 0; mi < 2; mi++)
#pragma unroll
      for (int ni = 0; ni < 4; ni++)
        acc[mi][ni] = wmma_f16f32(af[mi], bf[ni], acc[mi][ni]);
    __syncthreads();
  }

  const int cg = (lane >> 4) << 3, cn = lane & 15;
#pragma unroll
  for (int mi = 0; mi < 2; mi++)
#pragma unroll
    for (int ni = 0; ni < 4; ni++)
#pragma unroll
      for (int v = 0; v < 8; v++) {
        int m = m0 + wm + mi * 16 + v + cg;
        int n = n0 + wn + ni * 16 + cn;
        P[(size_t)m * N + n] = (_Float16)acc[mi][ni][v];
      }
}

// ---------------------------------------------------------------- kernel 2
// delta_mem[d][e] = sum_t (elu(k[t][d])+1) * v[t][e], delta_z[d] = sum_t ak
__global__ __launch_bounds__(256) void seg_delta_kernel(
    const _Float16* __restrict__ Kh, const _Float16* __restrict__ Vh,
    float* __restrict__ dmem, float* __restrict__ dz)
{
  extern __shared__ char smem[];
  _Float16* akT = (_Float16*)smem;        // [64][520]  A as [d][t]
  _Float16* vT  = akT + 64 * 520;         // [64][520]  B as [e][t]
  float*    zsh = (float*)(vT + 64 * 520);// [64]
  float*   dacc = zsh + 64;               // [64][64]

  const int s = blockIdx.x, h = blockIdx.y, b = blockIdx.z;
  const int tid = threadIdx.x, lane = tid & 31, wid = tid >> 5;
  const size_t rowbase = (size_t)b * L_ + (size_t)s * SEG;
  const int colbase = h * DK_;

  for (int i = tid; i < 64; i += 256) zsh[i] = 0.f;
  for (int i = tid; i < 4096; i += 256) dacc[i] = 0.f;
  __syncthreads();

  const int c0 = (tid & 7) << 3;
  float zp[8] = {0.f, 0.f, 0.f, 0.f, 0.f, 0.f, 0.f, 0.f};
  for (int i = 0; i < 16; i++) {
    int idx = tid + (i << 8);
    int t = idx >> 3;
    v8h kv = *(const v8h*)(Kh + (rowbase + t) * D_ + colbase + c0);
    v8h vv = *(const v8h*)(Vh + (rowbase + t) * D_ + colbase + c0);
#pragma unroll
    for (int j = 0; j < 8; j++) {
      float kf = (float)kv[j];
      float ak = kf > 0.f ? (kf + 1.f) : __expf(kf);  // elu(x)+1
      akT[(size_t)(c0 + j) * 520 + t] = (_Float16)ak;
      vT [(size_t)(c0 + j) * 520 + t] = vv[j];
      zp[j] += ak;
    }
  }
#pragma unroll
  for (int j = 0; j < 8; j++) atomicAdd(&zsh[c0 + j], zp[j]);
  __syncthreads();

  // wave wid owns K-range t in [wid*64, wid*64+64)
  const int tb = wid << 6;
  const v8f vz = {0.f, 0.f, 0.f, 0.f, 0.f, 0.f, 0.f, 0.f};
  v8f acc[4][4];
  for (int i = 0; i < 4; i++)
    for (int j = 0; j < 4; j++) acc[i][j] = vz;
#pragma unroll
  for (int kk = 0; kk < 64; kk += 32) {
    v16h af[4], bf[4];
#pragma unroll
    for (int mi = 0; mi < 4; mi++)
      af[mi] = ld_a_frag(akT + (size_t)(mi * 16) * 520 + tb + kk, 520, lane);
#pragma unroll
    for (int ni = 0; ni < 4; ni++)
      bf[ni] = ld_b_frag(vT + (size_t)(ni * 16) * 520 + tb + kk, 520, lane);
#pragma unroll
    for (int mi = 0; mi < 4; mi++)
#pragma unroll
      for (int ni = 0; ni < 4; ni++)
        acc[mi][ni] = wmma_f16f32(af[mi], bf[ni], acc[mi][ni]);
  }
  const int cg = (lane >> 4) << 3, cn = lane & 15;
#pragma unroll
  for (int mi = 0; mi < 4; mi++)
#pragma unroll
    for (int ni = 0; ni < 4; ni++)
#pragma unroll
      for (int v = 0; v < 8; v++)
        atomicAdd(&dacc[(mi * 16 + v + cg) * 64 + ni * 16 + cn],
                  acc[mi][ni][v]);
  __syncthreads();

  const size_t obase = (((size_t)(b * H_ + h)) * NSEG + s) * 4096;
  for (int i = tid; i < 4096; i += 256) dmem[obase + i] = dacc[i];
  const size_t zb = (((size_t)(b * H_ + h)) * NSEG + s) * DK_;
  for (int i = tid; i < 64; i += 256) dz[zb + i] = zsh[i];
}

// ---------------------------------------------------------------- kernel 3
__global__ __launch_bounds__(256) void prefix_kernel(
    const float* __restrict__ dmem, const float* __restrict__ dz,
    float* __restrict__ pmem, float* __restrict__ pz)
{
  const int bh = blockIdx.x, tid = threadIdx.x;
  const size_t base = (size_t)bh * NSEG * 4096;
  for (int e = tid; e < 4096; e += 256) {
    float run = 0.f;
    for (int s = 0; s < NSEG; s++) {
      pmem[base + (size_t)s * 4096 + e] = run;   // exclusive prefix (old state)
      run += dmem[base + (size_t)s * 4096 + e];
    }
  }
  const size_t zb = (size_t)bh * NSEG * DK_;
  for (int e = tid; e < DK_; e += 256) {
    float run = 0.f;
    for (int s = 0; s < NSEG; s++) {
      pz[zb + (size_t)s * DK_ + e] = run;
      run += dz[zb + (size_t)s * DK_ + e];
    }
  }
}

// ---------------------------------------------------------------- kernel 4
__global__ __launch_bounds__(256) void attn_kernel(
    const _Float16* __restrict__ Qh, const _Float16* __restrict__ Kh,
    const _Float16* __restrict__ Vh, const int* __restrict__ mask,
    const float* __restrict__ pmem, const float* __restrict__ pz,
    const float* __restrict__ betas, _Float16* __restrict__ Att)
{
  extern __shared__ char smem[];
  _Float16* k_s  = (_Float16*)smem;       // [512][72]   K natural (B of scores)
  _Float16* vT   = k_s + 512 * 72;        // [64][520]   V^T (B of A_dot)
  _Float16* q_s  = vT + 64 * 520;         // [32][72]
  _Float16* aq_s = q_s + 32 * 72;         // [32][72]
  _Float16* memT = aq_s + 32 * 72;        // [64][64]    mem^T (B of A_mem)
  _Float16* P16  = memT + 64 * 64;        // [32][520]   exp(scores) f16
  float* sc      = (float*)(P16 + 32 * 520); // [32][512]
  float* maskadd = sc + 32 * 512;         // [512]
  float* zs      = maskadd + 512;         // [64]
  float* gs      = zs + 64;               // [64]
  float* rowsum  = gs + 64;               // [32]
  float* denom   = rowsum + 32;           // [32]

  const int s = blockIdx.x, bh = blockIdx.y;
  const int b = bh >> 4, h = bh & 15;
  const int tid = threadIdx.x, lane = tid & 31, wid = tid >> 5;
  const size_t rowbase = (size_t)b * L_ + (size_t)s * SEG;
  const int colbase = h * DK_;
  const float inv_sqrt_dk = 0.125f;
  const v8f vz = {0.f, 0.f, 0.f, 0.f, 0.f, 0.f, 0.f, 0.f};

  // ---- stage segment K (async direct-to-LDS), V^T, mask, old mem/z, gate
  {
    const int c0 = (tid & 7) << 3;
    for (int i = 0; i < 16; i++) {
      int idx = tid + (i << 8);
      int t = idx >> 3;
      async_g2l_16B(Kh + (rowbase + t) * D_ + colbase + c0,
                    &k_s[(size_t)t * 72 + c0]);
      v8h vv = *(const v8h*)(Vh + (rowbase + t) * D_ + colbase + c0);
#pragma unroll
      for (int j = 0; j < 8; j++) vT[(size_t)(c0 + j) * 520 + t] = vv[j];
    }
  }
  for (int i = tid; i < SEG; i += 256)
    maskadd[i] = (mask[(size_t)b * L_ + (size_t)s * SEG + i] == 0) ? -1e9f : 0.f;
  {
    const size_t mb = ((size_t)bh * NSEG + s) * 4096;
    for (int i = tid; i < 4096; i += 256) {
      int d = i >> 6, e = i & 63;
      memT[e * 64 + d] = (_Float16)pmem[mb + i];
    }
    const size_t zb = ((size_t)bh * NSEG + s) * DK_;
    for (int i = tid; i < DK_; i += 256) {
      zs[i] = pz[zb + i];
      gs[i] = 1.f / (1.f + __expf(-betas[h * DK_ + i]));  // sigmoid(beta)
    }
  }
  wait_async0();
  __syncthreads();

  // ---- 16 query sub-blocks of 32 rows, reusing staged K/V
  for (int qb = 0; qb < 16; qb++) {
    const int qrow0 = s * SEG + qb * 32;  // local row within batch b

    {  // load q tile + aq = elu(q)+1
      int t = tid >> 3, c = (tid & 7) << 3;
      v8h qv = *(const v8h*)(Qh + ((size_t)b * L_ + qrow0 + t) * D_ + colbase + c);
      *(v8h*)&q_s[t * 72 + c] = qv;
#pragma unroll
      for (int j = 0; j < 8; j++) {
        float qf = (float)qv[j];
        float aq = qf > 0.f ? (qf + 1.f) : __expf(qf);
        aq_s[t * 72 + c + j] = (_Float16)aq;
      }
    }
    __syncthreads();

    {  // scores: wave wid owns key columns [wid*64, wid*64+64)
      const int nb = wid << 6;
      v8f sacc[2][4];
      for (int i = 0; i < 2; i++)
        for (int j = 0; j < 4; j++) sacc[i][j] = vz;
#pragma unroll
      for (int kk = 0; kk < 64; kk += 32) {
        v16h af[2], bf[4];
#pragma unroll
        for (int mi = 0; mi < 2; mi++)
          af[mi] = ld_a_frag(q_s + (size_t)(mi * 16) * 72 + kk, 72, lane);
#pragma unroll
        for (int ni = 0; ni < 4; ni++)
          bf[ni] = ld_b_frag(k_s + (size_t)(nb + ni * 16) * 72 + kk, 72, lane);
#pragma unroll
        for (int mi = 0; mi < 2; mi++)
#pragma unroll
          for (int ni = 0; ni < 4; ni++)
            sacc[mi][ni] = wmma_f16f32(af[mi], bf[ni], sacc[mi][ni]);
      }
      const int cg = (lane >> 4) << 3, cn = lane & 15;
#pragma unroll
      for (int mi = 0; mi < 2; mi++)
#pragma unroll
        for (int ni = 0; ni < 4; ni++)
#pragma unroll
          for (int v = 0; v < 8; v++) {
            int m = mi * 16 + v + cg;
            int n = nb + ni * 16 + cn;
            sc[m * 512 + n] = sacc[mi][ni][v] * inv_sqrt_dk + maskadd[n];
          }
    }
    __syncthreads();

    {  // softmax (unnormalized exp) + memory denominator; 8 lanes per row
      int r = tid >> 3, g = tid & 7;
      float mx = -3.4e38f;
      for (int c = g; c < SEG; c += 8) mx = fmaxf(mx, sc[r * 512 + c]);
#pragma unroll
      for (int off = 4; off >= 1; off >>= 1) mx = fmaxf(mx, __shfl_xor(mx, off, 8));
      float sum = 0.f;
      for (int c = g; c < SEG; c += 8) {
        float e = __expf(sc[r * 512 + c] - mx);
        sum += e;
        P16[r * 520 + c] = (_Float16)e;
      }
#pragma unroll
      for (int off = 4; off >= 1; off >>= 1) sum += __shfl_xor(sum, off, 8);
      if (g == 0) rowsum[r] = sum;
      float dp = 0.f;
      for (int d = g; d < DK_; d += 8) dp += (float)aq_s[r * 72 + d] * zs[d];
#pragma unroll
      for (int off = 4; off >= 1; off >>= 1) dp += __shfl_xor(dp, off, 8);
      if (g == 0) denom[r] = dp + 1e-9f;
    }
    __syncthreads();

    {  // A_dot = P @ V  and  A_mem = aq @ mem; one 16x16 frag per wave
      const int m0f = (wid & 1) << 4, n0f = (wid >> 1) << 4;
      v8f adot = vz, amem = vz;
      for (int kk = 0; kk < SEG; kk += 32) {
        v16h af = ld_a_frag(P16 + (size_t)m0f * 520 + kk, 520, lane);
        v16h bf = ld_b_frag(vT + (size_t)n0f * 520 + kk, 520, lane);
        adot = wmma_f16f32(af, bf, adot);
      }
#pragma unroll
      for (int kk = 0; kk < 64; kk += 32) {
        v16h af = ld_a_frag(aq_s + (size_t)m0f * 72 + kk, 72, lane);
        v16h bf = ld_b_frag(memT + (size_t)n0f * 64 + kk, 64, lane);
        amem = wmma_f16f32(af, bf, amem);
      }
      const int cg = (lane >> 4) << 3, cn = lane & 15;
#pragma unroll
      for (int v = 0; v < 8; v++) {
        int r = m0f + v + cg;
        int e = n0f + cn;
        float g = gs[e];
        float av = g * (amem[v] / denom[r]) + (1.f - g) * (adot[v] / rowsum[r]);
        Att[((size_t)b * L_ + qrow0 + r) * D_ + colbase + e] = (_Float16)av;
      }
    }
    __syncthreads();
  }
}

// ---------------------------------------------------------------- kernel 5
__global__ __launch_bounds__(256) void gemm_out_kernel(
    const _Float16* __restrict__ A, const _Float16* __restrict__ Wo,
    const float* __restrict__ bias, float* __restrict__ Out)
{
  __shared__ _Float16 As[GTM][GLDK];
  __shared__ _Float16 Bs[GTN][GLDK];
  const int N = D_, K = D_;
  const int m0 = blockIdx.y * GTM, n0 = blockIdx.x * GTN;
  const int tid = threadIdx.x, lane = tid & 31, wid = tid >> 5;
  const int wm = (wid & 3) * 32, wn = (wid >> 2) * 64;

  const v8f vz = {0.f, 0.f, 0.f, 0.f, 0.f, 0.f, 0.f, 0.f};
  v8f acc[2][4];
  for (int i = 0; i < 2; i++)
    for (int j = 0; j < 4; j++) acc[i][j] = vz;

  for (int k0 = 0; k0 < K; k0 += GTK) {
#pragma unroll
    for (int i = 0; i < 2; i++) {   // 128x32 f16 tiles: 512 x 16B each
      int idx = tid + (i << 8);
      int r = idx >> 2, c = (idx & 3) << 3;
      async_g2l_16B(A  + (size_t)(m0 + r) * K + k0 + c, &As[r][c]);
      async_g2l_16B(Wo + (size_t)(n0 + r) * K + k0 + c, &Bs[r][c]);
    }
    if (k0 + GTK < K)
      __builtin_prefetch(A + (size_t)(m0 + (tid >> 1)) * K + k0 + GTK, 0, 1);
    wait_async0();
    __syncthreads();

    v16h af[2], bf[4];
#pragma unroll
    for (int mi = 0; mi < 2; mi++)
      af[mi] = ld_a_frag(&As[wm + mi * 16][0], GLDK, lane);
#pragma unroll
    for (int ni = 0; ni < 4; ni++)
      bf[ni] = ld_b_frag(&Bs[wn + ni * 16][0], GLDK, lane);
#pragma unroll
    for (int mi = 0; mi < 2; mi++)
#pragma unroll
      for (int ni = 0; ni < 4; ni++)
        acc[mi][ni] = wmma_f16f32(af[mi], bf[ni], acc[mi][ni]);
    __syncthreads();
  }

  const int cg = (lane >> 4) << 3, cn = lane & 15;
#pragma unroll
  for (int mi = 0; mi < 2; mi++)
#pragma unroll
    for (int ni = 0; ni < 4; ni++)
#pragma unroll
      for (int v = 0; v < 8; v++) {
        int m = m0 + wm + mi * 16 + v + cg;
        int n = n0 + wn + ni * 16 + cn;
        Out[(size_t)m * N + n] = acc[mi][ni][v] + bias[n];
      }
}

// ---------------------------------------------------------------- launcher
extern "C" void kernel_launch(void* const* d_in, const int* in_sizes, int n_in,
                              void* d_out, int out_size, void* d_ws,
                              size_t ws_size, hipStream_t stream)
{
  (void)in_sizes; (void)n_in; (void)out_size; (void)ws_size;
  const float* x     = (const float*)d_in[0];
  const int*   mask  = (const int*)d_in[1];
  const float* wq    = (const float*)d_in[2];
  const float* wk    = (const float*)d_in[3];
  const float* wv    = (const float*)d_in[4];
  const float* wo_w  = (const float*)d_in[5];
  const float* wo_b  = (const float*)d_in[6];
  const float* betas = (const float*)d_in[7];
  float* out = (float*)d_out;

  const size_t MN = (size_t)B_ * L_ * D_;          // 33,554,432 elements
  const size_t WN = (size_t)D_ * D_;               // 1,048,576 elements
  _Float16* Qh   = (_Float16*)d_ws;
  _Float16* Kh   = Qh + MN;
  _Float16* Vh   = Kh + MN;
  _Float16* Att  = Vh + MN;
  _Float16* Xh   = Att + MN;
  _Float16* Wq16 = Xh + MN;
  _Float16* Wk16 = Wq16 + WN;
  _Float16* Wv16 = Wk16 + WN;
  _Float16* Wo16 = Wv16 + WN;
  float* dmem = (float*)(Wo16 + WN);
  float* pmem = dmem + (size_t)B_ * H_ * NSEG * DK_ * DK_;
  float* dz   = pmem + (size_t)B_ * H_ * NSEG * DK_ * DK_;
  float* pz   = dz   + (size_t)B_ * H_ * NSEG * DK_;

  const dim3 blk(256);
  const size_t lds_seg = (size_t)(64 * 520 * 2) * 2 + 64 * 4 + 64 * 64 * 4;
  const size_t lds_att =
      (size_t)(512 * 72 + 64 * 520 + 32 * 72 + 32 * 72 + 64 * 64 + 32 * 520) * 2 +
      (size_t)(32 * 512 + 512 + 64 + 64 + 32 + 32) * 4;

  const int n4x = (int)(MN / 4), n4w = (int)(WN / 4);
  conv_f16_kernel<<<dim3((n4x + 255) / 256), blk, 0, stream>>>(x, Xh, n4x);
  conv_f16_kernel<<<dim3((n4w + 255) / 256), blk, 0, stream>>>(wq, Wq16, n4w);
  conv_f16_kernel<<<dim3((n4w + 255) / 256), blk, 0, stream>>>(wk, Wk16, n4w);
  conv_f16_kernel<<<dim3((n4w + 255) / 256), blk, 0, stream>>>(wv, Wv16, n4w);
  conv_f16_kernel<<<dim3((n4w + 255) / 256), blk, 0, stream>>>(wo_w, Wo16, n4w);

  gemm_qkv_kernel<<<dim3(D_ / GTN, (B_ * L_) / GTM, 3), blk, 0, stream>>>(
      Xh, Wq16, Wk16, Wv16, Qh, Kh, Vh);
  seg_delta_kernel<<<dim3(NSEG, H_, B_), blk, lds_seg, stream>>>(Kh, Vh, dmem, dz);
  prefix_kernel<<<dim3(B_ * H_), blk, 0, stream>>>(dmem, dz, pmem, pz);
  attn_kernel<<<dim3(NSEG, B_ * H_), blk, lds_att, stream>>>(
      Qh, Kh, Vh, mask, pmem, pz, betas, Att);
  gemm_out_kernel<<<dim3(D_ / GTN, (B_ * L_) / GTM), blk, 0, stream>>>(
      Att, Wo16, wo_b, out);
}
